// SSM_56934086475819
// MI455X (gfx1250) — matrix operand order repaired and checked by reading, not compiled, not run on hardware
//
#include <hip/hip_runtime.h>
#include <hip/hip_bf16.h>
#include <math.h>

// ---------------------------------------------------------------------------
// SSM scan for MI455X (gfx1250, wave32, WMMA + TDM).
//   X[t] = X[t-1] @ W_eff + inj[t],  W_eff = (s * softmax(A_w, axis=1))^T
//   inj  = U @ B^T + D @ E^T
// Design:
//   * prep_w:   softmax+scale, packed directly into f32 WMMA B-fragment order
//   * pack_be:  B/E weights -> bf16 WMMA B-fragment order
//   * inj:      bf16 WMMA 16x16x32 (f32 accum) over K=128(+64), writes X
//   * scan:     persistent kernel, 64 blocks, grid barrier per step,
//               f32 WMMA 16x16x4 (full-precision recurrence),
//               x_prev strip staged into LDS by the Tensor Data Mover
//               (tensor_load_to_lds + s_wait_tensorcnt), HW LDS padding
//               (256 dwords -> +4 dwords) to spread ds_load_b64 banks.
// ---------------------------------------------------------------------------

typedef __attribute__((ext_vector_type(16))) __bf16 v16bf;
typedef __attribute__((ext_vector_type(8)))  float  v8f;
typedef __attribute__((ext_vector_type(2)))  float  v2f;
typedef __attribute__((ext_vector_type(4)))  unsigned u32x4;
typedef __attribute__((ext_vector_type(8)))  int      i32x8;
typedef __attribute__((ext_vector_type(4)))  int      i32x4;

#define NX 512
#define NU 128
#define ND 64
#define BATCH 256
#define T_STEPS 256
#define R_ROWS (T_STEPS * BATCH)   // 65536 rows of inj
#define NTILES_N (NX / 16)         // 32
#define SCAN_BLOCKS 64

// LDS layout for the x_prev strip matches TDM padding: after every 256 dwords
// of payload, 4 dwords of pad => 520-dword row stride (rows land on distinct
// bank groups: row*520 % 64 = row*8).
__device__ __forceinline__ int lds_off(int r, int c) {  // dword index
  return (r * 2 + (c >> 8)) * 260 + (c & 255);
}

#if defined(__has_builtin)
#if __has_builtin(__builtin_amdgcn_tensor_load_to_lds)
#define HAVE_TDM 1
#endif
#endif

// ---------------- W_eff -> f32 B-fragment pack -----------------------------
// B-frag (4x16 f32, 16x16x4): lane L: col n = L%16; vgpr v: k = tk*4 + (L<16?0:2) + v
__global__ void prep_w_kernel(const float* __restrict__ Aw,
                              const float* __restrict__ Asc,
                              float* __restrict__ Wpack) {
  __shared__ float red[256];
  const int i = blockIdx.x;           // row of M; M[i][j] -> Bfrag[k=j][n=i]
  const int tid = threadIdx.x;
  float a0 = Aw[i * NX + tid];
  float a1 = Aw[i * NX + tid + 256];
  red[tid] = fmaxf(a0, a1);
  __syncthreads();
  for (int s = 128; s > 0; s >>= 1) {
    if (tid < s) red[tid] = fmaxf(red[tid], red[tid + s]);
    __syncthreads();
  }
  const float mx = red[0];
  __syncthreads();
  const float e0 = expf(a0 - mx), e1 = expf(a1 - mx);
  red[tid] = e0 + e1;
  __syncthreads();
  for (int s = 128; s > 0; s >>= 1) {
    if (tid < s) red[tid] += red[tid + s];
    __syncthreads();
  }
  const float inv = 1.0f / red[0];
  const int nt = i >> 4;
  const int lanebase = i & 15;
#pragma unroll
  for (int h = 0; h < 2; ++h) {
    const int j = tid + h * 256;                 // k index
    const float sg = 1.0f / (1.0f + expf(-Asc[i * NX + j]));
    const float val = (h ? e1 : e0) * inv * (1.0f - 0.1f * sg);
    const int tk = j >> 2, rr = j & 3;
    const int lane = lanebase + ((rr >= 2) ? 16 : 0);
    const int v = rr & 1;
    Wpack[(((tk * NTILES_N + nt) * 32 + lane) << 1) + v] = val;
  }
}

// ---------------- B_weight / E_weight -> bf16 B-fragment pack --------------
// bf16 B-frag (32x16): lane L: col n = L%16; elem h: k = tk*32 + (L<16?0:8) + (h&7) + (h>=8?16:0)
__global__ void pack_be_kernel(const float* __restrict__ src,  // [NX][K] row-major
                               __bf16* __restrict__ dst, int K) {
  const int gid = blockIdx.x * blockDim.x + threadIdx.x;
  const int h = gid & 15;
  const int lane = (gid >> 4) & 31;
  const int nt = (gid >> 9) & 31;
  const int tk = gid >> 14;
  const int n = nt * 16 + (lane & 15);
  const int k = tk * 32 + ((lane < 16) ? 0 : 8) + (h & 7) + ((h >> 3) << 4);
  dst[gid] = (__bf16)src[n * K + k];
}

// bf16 A-frag (16x32): lane L: row = L%16; lo = (L<16?0:8);
__device__ __forceinline__ v16bf load_a_bf16(const float* __restrict__ p, int lo) {
  v16bf a;
#pragma unroll
  for (int v = 0; v < 4; ++v) {
    a[2 * v]     = (__bf16)p[lo + 2 * v];
    a[2 * v + 1] = (__bf16)p[lo + 2 * v + 1];
    a[8 + 2 * v]     = (__bf16)p[16 + lo + 2 * v];
    a[8 + 2 * v + 1] = (__bf16)p[16 + lo + 2 * v + 1];
  }
  return a;
}

// ---------------- injection GEMM: X[t] <- U@B^T + D@E^T (bf16 WMMA) --------
__global__ void inj_kernel(const float* __restrict__ U, const float* __restrict__ D,
                           const __bf16* __restrict__ Bpack,
                           const __bf16* __restrict__ Epack,
                           float* __restrict__ X) {
  const int lane = threadIdx.x & 31;
  const int wave = threadIdx.x >> 5;
  const int tile = blockIdx.x * 8 + wave;   // 131072 (4096 M-tiles x 32 N-tiles)
  const int mt = tile >> 5;
  const int nt = tile & 31;
  const int arow = mt * 16 + (lane & 15);
  const int lo = (lane < 16) ? 0 : 8;
  const v16bf* Bp = reinterpret_cast<const v16bf*>(Bpack);
  const v16bf* Ep = reinterpret_cast<const v16bf*>(Epack);
  v8f c = {};
#pragma unroll
  for (int tk = 0; tk < NU / 32; ++tk) {
    v16bf a = load_a_bf16(U + (size_t)arow * NU + tk * 32, lo);
    v16bf b = Bp[(tk * NTILES_N + nt) * 32 + lane];
    c = __builtin_amdgcn_wmma_f32_16x16x32_bf16(false, a, false, b, (short)0, c,
                                                false, false);
  }
#pragma unroll
  for (int tk = 0; tk < ND / 32; ++tk) {
    v16bf a = load_a_bf16(D + (size_t)arow * ND + tk * 32, lo);
    v16bf b = Ep[(tk * NTILES_N + nt) * 32 + lane];
    c = __builtin_amdgcn_wmma_f32_16x16x32_bf16(false, a, false, b, (short)0, c,
                                                false, false);
  }
  const int col = nt * 16 + (lane & 15);
  const int rbase = mt * 16 + ((lane < 16) ? 0 : 8);
#pragma unroll
  for (int r = 0; r < 8; ++r) X[(size_t)(rbase + r) * NX + col] = c[r];
}

__global__ void init_bar_kernel(unsigned* bar) {
  if (threadIdx.x < 2) bar[threadIdx.x] = 0u;
}

// ---------------- persistent scan: f32 WMMA 16x16x4, TDM stage, grid bar ---
__global__ void __launch_bounds__(256, 1)
scan_kernel(const float* __restrict__ x0, const float* __restrict__ Wpack,
            float* __restrict__ X, float* __restrict__ Y,
            unsigned* __restrict__ bar) {
  __shared__ float As[16 * 520];            // padded strip (TDM pad layout), 33.3 KB
  const int tid = threadIdx.x;
  const int lane = tid & 31;
  const int wave = tid >> 5;
  const int mtile = blockIdx.x >> 2;        // 0..15 (batch tiles)
  const int nstrip = blockIdx.x & 3;
  const int ntile = nstrip * 8 + wave;      // 0..31 (x-out tiles)
  const int arow = lane & 15;
  const int k0off = (lane < 16) ? 0 : 2;    // f32 A-frag: lanes16-31 hold K=2,3
  const int col = ntile * 16 + (lane & 15);
  const int rbase = mtile * 16 + ((lane < 16) ? 0 : 8);
  const v2f* Wp = reinterpret_cast<const v2f*>(Wpack);

#ifdef HAVE_TDM
  // Generic LDS pointer's low 32 bits == workgroup-relative LDS byte offset.
  const unsigned lds_base = (unsigned)(unsigned long long)(uintptr_t)(&As[0]);
  // D# group1 (constant): data_size=4B(2), pad_enable, pad_interval=7 (256 dw),
  // pad_amount=3 (4 dw); tensor_dim0=512, tensor_dim1=65536, tile=512x16,
  // tensor_dim0_stride=512.
  const i32x8 g1 = {(int)((2u << 16) | (1u << 20) | (7u << 22) | (3u << 25)),
                    (int)(512u << 16),          // tensor_dim0[15:0] @bits63:48
                    0,                          // tensor_dim0[31:16], dim1[15:0]
                    (int)(1u | (512u << 16)),   // dim1[31:16]=1, tile_dim0=512
                    16,                         // tile_dim1=16
                    512,                        // tensor_dim0_stride[31:0]
                    0, 0};
  const i32x4 gz = {0, 0, 0, 0};
#endif

  for (int t = 0; t < T_STEPS; ++t) {
    const float* src = (t == 0) ? x0 : (X + (size_t)(t - 1) * BATCH * NX);
    const float* srow = src + (size_t)(mtile * 16) * NX;

#ifdef HAVE_TDM
    if (wave == 0) {
      const unsigned long long ga = (unsigned long long)(uintptr_t)srow;
      u32x4 g0 = {1u,                                   // count=1 (valid D#)
                  lds_base,                             // lds_addr
                  (unsigned)(ga & 0xFFFFFFFFull),       // global_addr[31:0]
                  (unsigned)((ga >> 32) & 0x01FFFFFFull) | (2u << 30)}; // +type=2
#if __clang_major__ >= 23
      i32x8 g1b = {0, 0, 0, 0, 0, 0, 0, 0};
      __builtin_amdgcn_tensor_load_to_lds(g0, g1, gz, gz, g1b, 0);
#else
      __builtin_amdgcn_tensor_load_to_lds(g0, g1, gz, gz, 0);
#endif
      __builtin_amdgcn_s_wait_tensorcnt(0);
    }
    __syncthreads();
#else
    for (int e = tid; e < 16 * NX; e += 256) {
      const int r = e >> 9, cc = e & 511;
      As[lds_off(r, cc)] = srow[(size_t)r * NX + cc];
    }
    __syncthreads();
#endif

    float* Xt = X + (size_t)t * BATCH * NX;
    v8f c;
#pragma unroll
    for (int r = 0; r < 8; ++r) c[r] = Xt[(size_t)(rbase + r) * NX + col]; // inj

    for (int tk = 0; tk < NX / 4; ++tk) {
      v2f a;
      const float* ap = &As[lds_off(arow, tk * 4 + k0off)];
      a[0] = ap[0];                          // ds_load_b64 (pair never crosses pad)
      a[1] = ap[1];
      v2f b = Wp[(tk * NTILES_N + ntile) * 32 + lane];   // coalesced b64
      c = __builtin_amdgcn_wmma_f32_16x16x4_f32(false, a, false, b, (short)0, c,
                                                false, false);
    }

#pragma unroll
    for (int r = 0; r < 8; ++r) Xt[(size_t)(rbase + r) * NX + col] = c[r];
    if (ntile == NTILES_N - 1 && (lane & 15) == 15) {     // col 511 -> Y
#pragma unroll
      for (int r = 0; r < 8; ++r) Y[t * BATCH + rbase + r] = c[r];
    }

    // ---- grid-wide sense barrier (agent scope) ----
    __threadfence();
    __syncthreads();
    if (tid == 0) {
      unsigned g = __hip_atomic_load(&bar[1], __ATOMIC_ACQUIRE,
                                     __HIP_MEMORY_SCOPE_AGENT);
      unsigned a = __hip_atomic_fetch_add(&bar[0], 1u, __ATOMIC_ACQ_REL,
                                          __HIP_MEMORY_SCOPE_AGENT);
      if (a == SCAN_BLOCKS - 1) {
        __hip_atomic_store(&bar[0], 0u, __ATOMIC_RELAXED,
                           __HIP_MEMORY_SCOPE_AGENT);
        __hip_atomic_store(&bar[1], g + 1u, __ATOMIC_RELEASE,
                           __HIP_MEMORY_SCOPE_AGENT);
      } else {
        while (__hip_atomic_load(&bar[1], __ATOMIC_ACQUIRE,
                                 __HIP_MEMORY_SCOPE_AGENT) == g)
          __builtin_amdgcn_s_sleep(2);
      }
    }
    __syncthreads();
    __builtin_amdgcn_fence(__ATOMIC_ACQUIRE, "agent");  // see peers' X[t]
  }
}

// ---------------------------------------------------------------------------
extern "C" void kernel_launch(void* const* d_in, const int* in_sizes, int n_in,
                              void* d_out, int out_size, void* d_ws,
                              size_t ws_size, hipStream_t stream) {
  (void)in_sizes; (void)n_in; (void)out_size; (void)ws_size;
  const float* x  = (const float*)d_in[0];
  const float* U  = (const float*)d_in[1];
  const float* D  = (const float*)d_in[2];
  const float* Aw = (const float*)d_in[3];
  const float* Asc= (const float*)d_in[4];
  const float* Bw = (const float*)d_in[5];
  const float* Ew = (const float*)d_in[6];
  float* X = (float*)d_out;                       // [T][B][NX]
  float* Y = X + (size_t)T_STEPS * BATCH * NX;    // [T][B]

  char* ws = (char*)d_ws;
  float*  Wpack = (float*)ws;                                 // 1 MiB
  __bf16* Bpack = (__bf16*)(ws + (1u << 20));                 // 128 KiB
  __bf16* Epack = (__bf16*)(ws + (1u << 20) + (1u << 17));    // 64 KiB
  unsigned* bar = (unsigned*)(ws + (1u << 20) + (1u << 17) + (1u << 16));

  prep_w_kernel<<<NX, 256, 0, stream>>>(Aw, Asc, Wpack);
  pack_be_kernel<<<(4 << 14) / 256, 256, 0, stream>>>(Bw, Bpack, NU);
  pack_be_kernel<<<(2 << 14) / 256, 256, 0, stream>>>(Ew, Epack, ND);
  inj_kernel<<<(R_ROWS / 16) * NTILES_N / 8, 256, 0, stream>>>(U, D, Bpack,
                                                               Epack, X);
  init_bar_kernel<<<1, 32, 0, stream>>>(bar);
  scan_kernel<<<SCAN_BLOCKS, 256, 0, stream>>>(x, Wpack, X, Y, bar);
}
